// Encoder_47760036331980
// MI455X (gfx1250) — matrix operand order repaired and checked by reading, not compiled, not run on hardware
//
#include <hip/hip_runtime.h>
#include <math.h>

// ---------------- model constants ----------------
#define LNUM  4
#define EDIM  512
#define HNUM  8
#define DHEAD 64
#define FDIM  2048
#define BSZ   2
#define SLEN  2048
#define SKLEN 1024

// ---------------- CDNA5 async-copy availability ----------------
#if __has_builtin(__builtin_amdgcn_global_load_async_to_lds_b128) && \
    __has_builtin(__builtin_amdgcn_s_wait_asynccnt)
#define USE_ASYNC 1
#else
#define USE_ASYNC 0
#endif

// ---------------- WMMA plumbing (gfx1250, wave32) ----------------
typedef __attribute__((ext_vector_type(16))) __bf16 bf16x16;
typedef __attribute__((ext_vector_type(8)))  float  f32x8;
typedef __attribute__((ext_vector_type(4)))  int    i32x4;

union FragU { bf16x16 v; unsigned u[8]; unsigned short s[16]; };

__device__ __forceinline__ unsigned short f2bf(float f) {
  unsigned u = __float_as_uint(f);
  u += 0x7FFFu + ((u >> 16) & 1u);     // round-to-nearest-even
  return (unsigned short)(u >> 16);
}
__device__ __forceinline__ unsigned pk2bf(float a, float b) {
  return (unsigned)f2bf(a) | ((unsigned)f2bf(b) << 16);
}

__device__ __forceinline__ f32x8 wmma_bf16(bf16x16 a, bf16x16 b, f32x8 c) {
  // D = A(16x32) * B(32x16) + C, fp32 accumulate
  return __builtin_amdgcn_wmma_f32_16x16x32_bf16(
      false, a, false, b, (short)0, c, false, false);
}

// A-fragment (16x32 bf16), base = element (0,0) of tile (LDS or global), ld in u16 (even).
// ISA: lanes 0-15 row m; VGPR i<4 -> K=2i+8*half, i>=4 -> K=16+2(i-4)+8*half.
__device__ __forceinline__ bf16x16 fragA(const unsigned short* base, int ld) {
  const int lane = threadIdx.x & 31;
  const int m = lane & 15, h = lane >> 4;
  const unsigned* row = (const unsigned*)(base + (size_t)m * ld);
  FragU r;
#pragma unroll
  for (int i = 0; i < 8; ++i) {
    int idx = (i & 3) + ((i >> 2) << 3) + (h << 2);   // u32 index of K-pair
    r.u[i] = row[idx];
  }
  return r.v;
}

// B-fragment (32x16 KxN) from tile stored transposed as [n][k]; ld in u16 (even).
// ISA: lane n = column; VGPR i<4 -> K=2i+8*half, i>=4 -> K=16+2(i-4)+8*half (same as A).
__device__ __forceinline__ bf16x16 fragB(const unsigned short* baseT, int ld) {
  const int lane = threadIdx.x & 31;
  const int n = lane & 15, h = lane >> 4;
  const unsigned* row = (const unsigned*)(baseT + (size_t)n * ld);
  FragU r;
#pragma unroll
  for (int i = 0; i < 8; ++i) r.u[i] = row[(i & 3) + ((i >> 2) << 3) + (h << 2)];
  return r.v;
}

// 16-byte global->LDS copy: async on CDNA5 (ASYNCcnt), vector copy fallback.
__device__ __forceinline__ void stage_copy(const unsigned short* g, unsigned short* l) {
#if USE_ASYNC
  typedef __attribute__((address_space(1))) i32x4 gv4i;   // global int4
  typedef __attribute__((address_space(3))) i32x4 lv4i;   // LDS int4
  __builtin_amdgcn_global_load_async_to_lds_b128((gv4i*)g, (lv4i*)l, 0, 0);
#else
  *(float4*)l = *(const float4*)g;
#endif
}

// ---------------- GEMM: out = A[M,K](bf16) @ WT[N,K]^T(bf16) + bias (+res)(+GELU) ----------
// Block 256 thr (8 waves, 2x4 grid of 64x64 wave tiles) -> 128x256 tile, K-step 32,
// double-buffered LDS fed by async b128 copies. 16 WMMAs per wave per stage.
__global__ __launch_bounds__(256) void gemm_kernel(
    const unsigned short* __restrict__ A, const unsigned short* __restrict__ WT,
    const float* __restrict__ bias, const float* __restrict__ R,
    float* __restrict__ Cf, unsigned short* __restrict__ Cb,
    int M, int N, int K, int act)
{
  // row stride 40 u16 = 80B: 16B-aligned for b128, m*20 mod 64 distinct -> no bank conflicts
  __shared__ __align__(16) unsigned short sAbuf[2][128 * 40];
  __shared__ __align__(16) unsigned short sBbuf[2][256 * 40];

  const int tid  = threadIdx.x;
  const int wave = tid >> 5, lane = tid & 31;
  const int wy = wave >> 2, wx = wave & 3;
  const int m0 = blockIdx.y << 7, n0 = blockIdx.x << 8;

  const unsigned short* Ab = A  + (size_t)m0 * K;
  const unsigned short* Wb = WT + (size_t)n0 * K;

  f32x8 acc[4][4];
#pragma unroll
  for (int a = 0; a < 4; ++a)
#pragma unroll
    for (int b = 0; b < 4; ++b) acc[a][b] = (f32x8)0.0f;

  // issue one stage of copies: A 128x32 (512 16B chunks), B 256x32 (1024 chunks)
  auto issue = [&](int k0, int buf) {
#pragma unroll
    for (int c = 0; c < 2; ++c) {
      int lin = tid + (c << 8);
      int row = lin >> 2, q = lin & 3;
      stage_copy(Ab + (size_t)row * K + k0 + (q << 3), &sAbuf[buf][row * 40 + (q << 3)]);
    }
#pragma unroll
    for (int c = 0; c < 4; ++c) {
      int lin = tid + (c << 8);
      int row = lin >> 2, q = lin & 3;
      stage_copy(Wb + (size_t)row * K + k0 + (q << 3), &sBbuf[buf][row * 40 + (q << 3)]);
    }
  };

  const int nst = K >> 5;
  issue(0, 0);
  for (int s = 0; s < nst; ++s) {
    const int cur = s & 1;
    if (s + 1 < nst) {
      issue((s + 1) << 5, cur ^ 1);
#if USE_ASYNC
      __builtin_amdgcn_s_wait_asynccnt(6);   // retire stage s (per-wave, in-order)
#endif
    } else {
#if USE_ASYNC
      __builtin_amdgcn_s_wait_asynccnt(0);
#endif
    }
    __syncthreads();

    bf16x16 af[4], bfr[4];
#pragma unroll
    for (int mi = 0; mi < 4; ++mi)
      af[mi] = fragA(&sAbuf[cur][((wy << 6) + (mi << 4)) * 40], 40);
#pragma unroll
    for (int ni = 0; ni < 4; ++ni)
      bfr[ni] = fragB(&sBbuf[cur][((wx << 6) + (ni << 4)) * 40], 40);
#pragma unroll
    for (int mi = 0; mi < 4; ++mi)
#pragma unroll
      for (int ni = 0; ni < 4; ++ni)
        acc[mi][ni] = wmma_bf16(af[mi], bfr[ni], acc[mi][ni]);
    __syncthreads();
  }

  // epilogue: bias (+residual) (+GELU-erf); fp32 and/or bf16 stores
  const int ml = lane & 15, hf = lane >> 4;
#pragma unroll
  for (int mi = 0; mi < 4; ++mi)
#pragma unroll
    for (int ni = 0; ni < 4; ++ni)
#pragma unroll
      for (int r = 0; r < 8; ++r) {
        int row = m0 + (wy << 6) + (mi << 4) + r + (hf << 3);
        int col = n0 + (wx << 6) + (ni << 4) + ml;
        float c = acc[mi][ni][r] + bias[col];
        if (R) c += R[(size_t)row * N + col];
        if (act) c = 0.5f * c * (1.0f + erff(c * 0.70710678118654752f));
        if (Cf) Cf[(size_t)row * N + col] = c;
        if (Cb) Cb[(size_t)row * N + col] = f2bf(c);
      }
}

// ---------------- Flash attention (bf16 Q/K/V/O, bf16 WMMA, online softmax) ----------------
// Grid: (Sq/64, H, B). Block 128 thr = 4 waves; each wave owns a 16-query tile, the
// block shares staged 32-key K/V tiles in LDS. Scale folded into softmax.
__global__ __launch_bounds__(128) void flash_kernel(
    const unsigned short* __restrict__ Q, const unsigned short* __restrict__ K,
    const unsigned short* __restrict__ V, unsigned short* __restrict__ O,
    int Sq, int Skv)
{
  const int b = blockIdx.z, h = blockIdx.y;
  const int tid = threadIdx.x;
  const int wave = tid >> 5, lane = tid & 31;
  const int ml = lane & 15, hf = lane >> 4;
  const int q0 = (blockIdx.x * 4 + wave) << 4;

  __shared__ unsigned shK[32 * 33];        // [key][d] bf16, stride 66 u16 (padded)
  __shared__ unsigned shVT[64 * 17];       // [d][key] bf16, stride 34 u16 (padded)
  __shared__ unsigned shP[4][16 * 17];     // per-wave probs [row][key], stride 34

  const unsigned short* Qb = Q + ((size_t)b * Sq + q0) * EDIM + h * DHEAD;
  const unsigned short* Kb = K + (size_t)b * Skv * EDIM + h * DHEAD;
  const unsigned short* Vb = V + (size_t)b * Skv * EDIM + h * DHEAD;

  const bf16x16 qa0 = fragA(Qb, EDIM);
  const bf16x16 qa1 = fragA(Qb + 32, EDIM);

  f32x8 o0 = (f32x8)0.0f, o1 = (f32x8)0.0f, o2 = (f32x8)0.0f, o3 = (f32x8)0.0f;
  float mrow[8], lrow[8];
#pragma unroll
  for (int r = 0; r < 8; ++r) { mrow[r] = -3.0e38f; lrow[r] = 0.0f; }

  unsigned short* const sK16  = (unsigned short*)shK;
  unsigned short* const sVT16 = (unsigned short*)shVT;
  unsigned short* const sP16  = (unsigned short*)shP[wave];

  for (int kt = 0; kt < Skv; kt += 32) {
    { // stage 32 keys x 64 dims: K row-major, V transposed
      int row = tid >> 2;            // 0..31 key
      int c0 = (tid & 3) << 4;       // d chunk of 16 (8 u32)
      const unsigned* kr = (const unsigned*)(Kb + (size_t)(kt + row) * EDIM + c0);
      const unsigned short* vr = Vb + (size_t)(kt + row) * EDIM + c0;
#pragma unroll
      for (int j = 0; j < 8; ++j) shK[row * 33 + (c0 >> 1) + j] = kr[j];
#pragma unroll
      for (int j = 0; j < 16; ++j) sVT16[(c0 + j) * 34 + row] = vr[j];
    }
    __syncthreads();

    // scores S = Q (16x64) @ K^T (64x32): two 16x16 tiles, contraction split 32+32
    f32x8 s0 = (f32x8)0.0f, s1 = (f32x8)0.0f;
    s0 = wmma_bf16(qa0, fragB(sK16, 66), s0);
    s0 = wmma_bf16(qa1, fragB(sK16 + 32, 66), s0);
    s1 = wmma_bf16(qa0, fragB(sK16 + 16 * 66, 66), s1);
    s1 = wmma_bf16(qa1, fragB(sK16 + 16 * 66 + 32, 66), s1);

    // online softmax in C-layout (each acc reg = one row); 16-lane half reductions
#pragma unroll
    for (int r = 0; r < 8; ++r) {
      float a0 = s0[r] * 0.125f, a1 = s1[r] * 0.125f;   // 1/sqrt(64)
      float x = fmaxf(a0, a1);
      x = fmaxf(x, __shfl_xor(x, 1, 32));
      x = fmaxf(x, __shfl_xor(x, 2, 32));
      x = fmaxf(x, __shfl_xor(x, 4, 32));
      x = fmaxf(x, __shfl_xor(x, 8, 32));
      float mn = fmaxf(mrow[r], x);
      float alpha = expf(mrow[r] - mn);
      mrow[r] = mn;
      float p0 = expf(a0 - mn);
      float p1 = expf(a1 - mn);
      float rs = p0 + p1;
      rs += __shfl_xor(rs, 1, 32);
      rs += __shfl_xor(rs, 2, 32);
      rs += __shfl_xor(rs, 4, 32);
      rs += __shfl_xor(rs, 8, 32);
      lrow[r] = lrow[r] * alpha + rs;
      o0[r] *= alpha; o1[r] *= alpha; o2[r] *= alpha; o3[r] *= alpha;
      int row = r + (hf << 3);
      sP16[row * 34 + ml]      = f2bf(p0);   // C-layout -> A-layout via LDS bounce
      sP16[row * 34 + 16 + ml] = f2bf(p1);
    }

    // O += P (16x32) @ V (32x64); same-wave LDS ops are in-order
    const bf16x16 pa = fragA(sP16, 34);
    o0 = wmma_bf16(pa, fragB(sVT16 + 0 * 16 * 34, 34), o0);
    o1 = wmma_bf16(pa, fragB(sVT16 + 1 * 16 * 34, 34), o1);
    o2 = wmma_bf16(pa, fragB(sVT16 + 2 * 16 * 34, 34), o2);
    o3 = wmma_bf16(pa, fragB(sVT16 + 3 * 16 * 34, 34), o3);
    __syncthreads();   // before next tile overwrites shK/shVT
  }

  unsigned short* Ob = O + ((size_t)b * Sq + q0) * EDIM + h * DHEAD;
#pragma unroll
  for (int r = 0; r < 8; ++r) {
    float inv = 1.0f / lrow[r];
    int row = r + (hf << 3);
    Ob[(size_t)row * EDIM +  0 + ml] = f2bf(o0[r] * inv);
    Ob[(size_t)row * EDIM + 16 + ml] = f2bf(o1[r] * inv);
    Ob[(size_t)row * EDIM + 32 + ml] = f2bf(o2[r] * inv);
    Ob[(size_t)row * EDIM + 48 + ml] = f2bf(o3[r] * inv);
  }
}

// ---------------- LayerNorm (ddof=1, eps on std) + optional bf16 mirror ----------------
__global__ __launch_bounds__(256) void ln_kernel(
    const float* __restrict__ X, const float* __restrict__ g,
    const float* __restrict__ bt, float* __restrict__ Y,
    unsigned short* __restrict__ Yb)
{
  const int row = blockIdx.x;
  const int t = threadIdx.x;
  __shared__ float red[256];
  const float* x = X + (size_t)row * EDIM;
  float v0 = x[t], v1 = x[t + 256];
  red[t] = v0 + v1;
  __syncthreads();
  for (int s = 128; s > 0; s >>= 1) { if (t < s) red[t] += red[t + s]; __syncthreads(); }
  float mean = red[0] * (1.0f / EDIM);
  __syncthreads();
  float d0 = v0 - mean, d1 = v1 - mean;
  red[t] = d0 * d0 + d1 * d1;
  __syncthreads();
  for (int s = 128; s > 0; s >>= 1) { if (t < s) red[t] += red[t + s]; __syncthreads(); }
  float var = red[0] * (1.0f / (EDIM - 1));
  float inv = 1.0f / (sqrtf(var) + 1e-6f);
  float y0 = g[t] * d0 * inv + bt[t];
  float y1 = g[t + 256] * d1 * inv + bt[t + 256];
  Y[(size_t)row * EDIM + t]       = y0;
  Y[(size_t)row * EDIM + t + 256] = y1;
  if (Yb) {
    Yb[(size_t)row * EDIM + t]       = f2bf(y0);
    Yb[(size_t)row * EDIM + t + 256] = f2bf(y1);
  }
}

// ---------------- weight transpose-convert: W[K][N] f32 -> WT[N][K] bf16 ----------------
__global__ __launch_bounds__(256) void wconv_kernel(
    const float* __restrict__ W, unsigned short* __restrict__ WT, int K, int N)
{
  __shared__ float t[32][33];
  const int n0 = blockIdx.x << 5, k0 = blockIdx.y << 5;
  const int tx = threadIdx.x & 31, ty = threadIdx.x >> 5;
#pragma unroll
  for (int i = ty; i < 32; i += 8)
    t[i][tx] = W[(size_t)(k0 + i) * N + n0 + tx];
  __syncthreads();
#pragma unroll
  for (int i = ty; i < 32; i += 8)
    WT[(size_t)(n0 + i) * K + k0 + tx] = f2bf(t[tx][i]);
}

// ---------------- fp32 -> bf16 (+ optional fp32 copy) ----------------
__global__ __launch_bounds__(256) void cvt_kernel(
    const float* __restrict__ X, float* __restrict__ Yf,
    unsigned short* __restrict__ Yb, int n4)
{
  int i = blockIdx.x * 256 + threadIdx.x;
  if (i >= n4) return;
  float4 v = ((const float4*)X)[i];
  if (Yf) ((float4*)Yf)[i] = v;
  unsigned long long p = (unsigned long long)pk2bf(v.x, v.y) |
                         ((unsigned long long)pk2bf(v.z, v.w) << 32);
  ((unsigned long long*)Yb)[i] = p;
}

// ---------------- driver ----------------
extern "C" void kernel_launch(void* const* d_in, const int* in_sizes, int n_in,
                              void* d_out, int out_size, void* d_ws, size_t ws_size,
                              hipStream_t stream)
{
  (void)in_sizes; (void)n_in; (void)out_size; (void)ws_size;
  const float* sen      = (const float*)d_in[0];
  const float* know     = (const float*)d_in[1];
  const float* sa_qkv_w = (const float*)d_in[2];
  const float* sa_qkv_b = (const float*)d_in[3];
  const float* sa_out_w = (const float*)d_in[4];
  const float* sa_out_b = (const float*)d_in[5];
  const float* ca_qkv_w = (const float*)d_in[6];
  const float* ca_qkv_b = (const float*)d_in[7];
  const float* ca_out_w = (const float*)d_in[8];
  const float* ca_out_b = (const float*)d_in[9];
  const float* ff_w1    = (const float*)d_in[10];
  const float* ff_b1    = (const float*)d_in[11];
  const float* ff_w2    = (const float*)d_in[12];
  const float* ff_b2    = (const float*)d_in[13];
  const float* ln_g     = (const float*)d_in[14];
  const float* ln_b     = (const float*)d_in[15];
  float* out = (float*)d_out;

  const int TOK  = BSZ * SLEN;                  // 4096
  const int TOKK = BSZ * SKLEN;                 // 2048
  const size_t NT = (size_t)TOK * EDIM;         // 2M elems
  const size_t EE = (size_t)EDIM * EDIM;
  const size_t EF = (size_t)EDIM * FDIM;
  const size_t WL = 8 * EE + 2 * EF;            // bf16 weight elems / layer

  // ws layout: fp32 [hid | f1 | f2] then bf16 region. Total ~100 MB.
  float* wsf = (float*)d_ws;
  float* hid = wsf;
  float* f1  = wsf + 1 * NT;
  float* f2  = wsf + 2 * NT;
  unsigned short* wsb = (unsigned short*)(wsf + 3 * NT);
  unsigned short* hid_b = wsb;
  unsigned short* actb  = wsb + 1 * NT;
  unsigned short* qb    = wsb + 2 * NT;
  unsigned short* kb    = wsb + 3 * NT;
  unsigned short* vb    = wsb + 4 * NT;
  unsigned short* ctxb  = wsb + 5 * NT;
  unsigned short* knowb = wsb + 6 * NT;                   // 1M used
  unsigned short* ffb   = wsb + 7 * NT;                   // 8M elems
  unsigned short* wtb   = wsb + 11 * NT;                  // 4*WL elems

  // ---- one-time (per launch) weight transpose-convert + input conversions ----
  auto T = [&](const float* Wsrc, unsigned short* dst, int K, int N) {
    wconv_kernel<<<dim3(N / 32, K / 32), 256, 0, stream>>>(Wsrc, dst, K, N);
  };
  for (int i = 0; i < LNUM; ++i) {
    unsigned short* lw = wtb + (size_t)i * WL;
    for (int j = 0; j < 3; ++j)
      T(sa_qkv_w + (size_t)i * 3 * EE + j * EE, lw + j * EE, EDIM, EDIM);
    T(sa_out_w + (size_t)i * EE, lw + 3 * EE, EDIM, EDIM);
    for (int j = 0; j < 3; ++j)
      T(ca_qkv_w + (size_t)i * 3 * EE + j * EE, lw + (4 + j) * EE, EDIM, EDIM);
    T(ca_out_w + (size_t)i * EE, lw + 7 * EE, EDIM, EDIM);
    T(ff_w1 + (size_t)i * EF, lw + 8 * EE, EDIM, FDIM);           // [F][E]
    T(ff_w2 + (size_t)i * EF, lw + 8 * EE + EF, FDIM, EDIM);      // [E][F]
  }
  cvt_kernel<<<dim3((unsigned)(NT / 4 / 256)), 256, 0, stream>>>(sen, hid, hid_b, (int)(NT / 4));
  cvt_kernel<<<dim3((unsigned)((size_t)TOKK * EDIM / 4 / 256)), 256, 0, stream>>>(
      know, nullptr, knowb, (int)((size_t)TOKK * EDIM / 4));

  const dim3 blk(256);
  const dim3 gEE(EDIM / 256, TOK / 128);     // 2 x 32
  const dim3 gEEk(EDIM / 256, TOKK / 128);   // 2 x 16
  const dim3 gEF(FDIM / 256, TOK / 128);     // 8 x 32
  const dim3 gAttn(SLEN / 64, HNUM, BSZ);    // 32 x 8 x 2
  const dim3 gLN(TOK);

  for (int i = 0; i < LNUM; ++i) {
    unsigned short* lw = wtb + (size_t)i * WL;
    const float* sb = sa_qkv_b + (size_t)i * 3 * EDIM;
    const float* cb = ca_qkv_b + (size_t)i * 3 * EDIM;
    const float* g  = ln_g + (size_t)i * EDIM;
    const float* be = ln_b + (size_t)i * EDIM;

    // ---- self-attention ----
    gemm_kernel<<<gEE, blk, 0, stream>>>(hid_b, lw + 0 * EE, sb,            nullptr, nullptr, qb, TOK, EDIM, EDIM, 0);
    gemm_kernel<<<gEE, blk, 0, stream>>>(hid_b, lw + 1 * EE, sb + EDIM,     nullptr, nullptr, kb, TOK, EDIM, EDIM, 0);
    gemm_kernel<<<gEE, blk, 0, stream>>>(hid_b, lw + 2 * EE, sb + 2 * EDIM, nullptr, nullptr, vb, TOK, EDIM, EDIM, 0);
    flash_kernel<<<gAttn, dim3(128), 0, stream>>>(qb, kb, vb, ctxb, SLEN, SLEN);
    gemm_kernel<<<gEE, blk, 0, stream>>>(ctxb, lw + 3 * EE, sa_out_b + (size_t)i * EDIM,
                                         hid, f1, nullptr, TOK, EDIM, EDIM, 0);
    ln_kernel<<<gLN, blk, 0, stream>>>(f1, g, be, f2, actb);                       // inter

    // ---- cross-attention ----
    gemm_kernel<<<gEE,  blk, 0, stream>>>(actb,  lw + 4 * EE, cb,            nullptr, nullptr, qb, TOK,  EDIM, EDIM, 0);
    gemm_kernel<<<gEEk, blk, 0, stream>>>(knowb, lw + 5 * EE, cb + EDIM,     nullptr, nullptr, kb, TOKK, EDIM, EDIM, 0);
    gemm_kernel<<<gEEk, blk, 0, stream>>>(knowb, lw + 6 * EE, cb + 2 * EDIM, nullptr, nullptr, vb, TOKK, EDIM, EDIM, 0);
    flash_kernel<<<gAttn, dim3(128), 0, stream>>>(qb, kb, vb, ctxb, SLEN, SKLEN);
    gemm_kernel<<<gEE, blk, 0, stream>>>(ctxb, lw + 7 * EE, ca_out_b + (size_t)i * EDIM,
                                         f2, f1, nullptr, TOK, EDIM, EDIM, 0);
    ln_kernel<<<gLN, blk, 0, stream>>>(f1, g, be, f2, actb);                       // co

    // ---- FFN (GELU-erf fused) ----
    gemm_kernel<<<gEF, blk, 0, stream>>>(actb, lw + 8 * EE, ff_b1 + (size_t)i * FDIM,
                                         nullptr, nullptr, ffb, TOK, FDIM, EDIM, 1);
    gemm_kernel<<<gEE, blk, 0, stream>>>(ffb, lw + 8 * EE + EF, ff_b2 + (size_t)i * EDIM,
                                         f2, f1, nullptr, TOK, EDIM, FDIM, 0);
    ln_kernel<<<gLN, blk, 0, stream>>>(f1, g, be, (i == LNUM - 1) ? out : hid,
                                       (i == LNUM - 1) ? nullptr : hid_b);
  }
}
